// CosineDeconf_58076547777061
// MI455X (gfx1250) — matrix operand order repaired and checked by reading, not compiled, not run on hardware
//
#include <hip/hip_runtime.h>
#include <math.h>

// ---------------------------------------------------------------------------
// Cosine similarity: out[n,c] = <x[n], w[c]> / (||x[n]|| * ||w[c]||)
//   X: [32768, 512] f32, W: [1000, 512] f32, out: [32768, 1000] f32
//
// GEMM on the bf16 WMMA pipe with split-precision (hi/lo) for ~fp32 accuracy:
//   x = hi + lo, hi = bf16(x), lo = bf16(x - hi)   (exact, bf16 exp == f32 exp)
//   dot ~= hi_a*hi_b + hi_a*lo_b + lo_a*hi_b       (3 bf16 WMMAs / tile)
//
// Double-buffered LDS ping-pong, batched global fetch into registers so the
// 8 global_load_b128 per K-iter issue back-to-back and retire under the WMMAs.
// ---------------------------------------------------------------------------

#define BATCH 32768
#define NCLS  1000
#define DFEAT 512

#define BM 128          // macro-tile M
#define BN 128          // macro-tile N
#define BK 32           // K tile (== WMMA bf16 K)
#define NT (DFEAT / BK) // 16 K-iterations
#define LDT 40          // padded LDS row stride (bf16 elems): 80B spreads banks

typedef __attribute__((ext_vector_type(16))) __bf16 v16bf;
typedef __attribute__((ext_vector_type(8)))  __bf16 v8bf;
typedef __attribute__((ext_vector_type(4)))  __bf16 v4bf;
typedef __attribute__((ext_vector_type(8)))  float  v8f;

// ---------------------------------------------------------------------------
// Reciprocal L2 norm per row (one wave32 per row).
// ---------------------------------------------------------------------------
__global__ __launch_bounds__(256)
void cosine_rnorm_kernel(const float* __restrict__ src,
                         float* __restrict__ dst, int nrows) {
  const int lane = threadIdx.x & 31;
  const int wave = threadIdx.x >> 5;
  const int row  = blockIdx.x * 8 + wave;
  if (row >= nrows) return;
  const float* p = src + (size_t)row * DFEAT;
  float s = 0.f;
#pragma unroll
  for (int j = 0; j < DFEAT / 32; ++j) {
    float v = p[lane + j * 32];
    s += v * v;
  }
#pragma unroll
  for (int off = 16; off > 0; off >>= 1) s += __shfl_xor(s, off, 32);
  if (lane == 0) dst[row] = rsqrtf(s);
}

// ---------------------------------------------------------------------------
// Helpers
// ---------------------------------------------------------------------------
__device__ __forceinline__ void split4(float4 v, v4bf& h, v4bf& l) {
  float f[4] = {v.x, v.y, v.z, v.w};
#pragma unroll
  for (int i = 0; i < 4; ++i) {
    __bf16 hi = (__bf16)f[i];
    h[i] = hi;
    l[i] = (__bf16)(f[i] - (float)hi);
  }
}

// A fragment (16x32 bf16, ISA §7.12.2): lane half 0 holds K {0..7, 16..23},
// lane half 1 holds K {8..15, 24..31}.  p = &tile[row*LDT + (lane>>4)*8].
__device__ __forceinline__ v16bf ld_frag_a(const __bf16* p) {
  v8bf c0 = *(const v8bf*)(p);        // K = k0    .. k0+7
  v8bf c1 = *(const v8bf*)(p + 16);   // K = k0+16 .. k0+23
  v16bf r;
#pragma unroll
  for (int i = 0; i < 8; ++i) { r[i] = c0[i]; r[i + 8] = c1[i]; }
  return r;
}

// B fragment (32x16 bf16): same K layout as A: lane half 0 holds K {0..7, 16..23},
// lane half 1 holds K {8..15, 24..31}.  p = &tile[row*LDT + (lane>>4)*8].
__device__ __forceinline__ v16bf ld_frag_b(const __bf16* p) {
  v8bf c0 = *(const v8bf*)(p);
  v8bf c1 = *(const v8bf*)(p + 16);
  v16bf r;
#pragma unroll
  for (int i = 0; i < 8; ++i) { r[i] = c0[i]; r[i + 8] = c1[i]; }
  return r;
}

__device__ __forceinline__ v8f wmma_bf16(v16bf a, v16bf b, v8f c) {
  // (neg_a, A, neg_b, B, c_mod, C, reuse_a, reuse_b)
  return __builtin_amdgcn_wmma_f32_16x16x32_bf16(false, a, false, b,
                                                 (short)0, c, false, false);
}

// ---------------------------------------------------------------------------
// Main fused GEMM + normalize kernel.
// 256 threads = 8 waves arranged 4(M) x 2(N); wave sub-tile = 32x64.
// ---------------------------------------------------------------------------
__global__ __launch_bounds__(256)
void cosine_gemm_kernel(const float* __restrict__ X, const float* __restrict__ W,
                        const float* __restrict__ xrn, const float* __restrict__ wrn,
                        float* __restrict__ out) {
  __shared__ __bf16 AsHi[2][BM * LDT];
  __shared__ __bf16 AsLo[2][BM * LDT];
  __shared__ __bf16 BsHi[2][BN * LDT];
  __shared__ __bf16 BsLo[2][BN * LDT];

  const int tid  = threadIdx.x;
  const int lane = tid & 31;
  const int wave = tid >> 5;
  const int wm   = wave & 3;            // wave row (0..3) -> M offset wm*32
  const int wn   = wave >> 2;           // wave col (0..1) -> N offset wn*64
  const int m0   = blockIdx.x * BM;
  const int n0   = blockIdx.y * BN;

  v8f acc[2][4];
#pragma unroll
  for (int mi = 0; mi < 2; ++mi)
#pragma unroll
    for (int ni = 0; ni < 4; ++ni)
#pragma unroll
      for (int j = 0; j < 8; ++j) acc[mi][ni][j] = 0.f;

  const int fr = lane & 15;             // fragment row within 16
  const int ka = (lane >> 4) * 8;       // A-fragment K base
  const int kb = (lane >> 4) * 8;       // B-fragment K base

  // Per-thread staging coordinates: 4 float4 per tile (f = tid + i*256).
  int srow[4], skv[4];
#pragma unroll
  for (int i = 0; i < 4; ++i) {
    const int f = tid + i * 256;        // 0..1023
    srow[i] = f >> 3;                   // tile row 0..127
    skv[i]  = (f & 7) << 2;             // float offset 0..28
  }

  float4 areg[4], breg[4];

  // ---- batched global fetch of K-tile kt into registers ----
  auto fetch = [&](int kt) {
    const float* Ab = X + (size_t)m0 * DFEAT + kt * BK;
    const float* Bb = W + kt * BK;
#pragma unroll
    for (int i = 0; i < 4; ++i)
      areg[i] = *(const float4*)(Ab + (size_t)srow[i] * DFEAT + skv[i]);
#pragma unroll
    for (int i = 0; i < 4; ++i) {
      const int cr = n0 + srow[i];
      breg[i] = (cr < NCLS) ? *(const float4*)(Bb + (size_t)cr * DFEAT + skv[i])
                            : make_float4(0.f, 0.f, 0.f, 0.f);
    }
  };

  // ---- split staged registers to bf16 hi/lo and store into LDS buffer b ----
  auto stage = [&](int b) {
#pragma unroll
    for (int i = 0; i < 4; ++i) {
      v4bf h, l;
      split4(areg[i], h, l);
      *(v4bf*)(AsHi[b] + srow[i] * LDT + skv[i]) = h;
      *(v4bf*)(AsLo[b] + srow[i] * LDT + skv[i]) = l;
      split4(breg[i], h, l);
      *(v4bf*)(BsHi[b] + srow[i] * LDT + skv[i]) = h;
      *(v4bf*)(BsLo[b] + srow[i] * LDT + skv[i]) = l;
    }
  };

  // ---- prologue: tile 0 into buffer 0 ----
  fetch(0);
  stage(0);
  __syncthreads();

  for (int kt = 0; kt < NT; ++kt) {
    const int buf = kt & 1;

    // Issue next tile's global loads; they retire under the WMMAs below.
    if (kt + 1 < NT) fetch(kt + 1);

    // ---- fragments from LDS[buf] ----
    v16bf aHi[2], aLo[2], bHi[4], bLo[4];
#pragma unroll
    for (int mi = 0; mi < 2; ++mi) {
      const int row = wm * 32 + mi * 16 + fr;
      aHi[mi] = ld_frag_a(AsHi[buf] + row * LDT + ka);
      aLo[mi] = ld_frag_a(AsLo[buf] + row * LDT + ka);
    }
#pragma unroll
    for (int ni = 0; ni < 4; ++ni) {
      const int row = wn * 64 + ni * 16 + fr;
      bHi[ni] = ld_frag_b(BsHi[buf] + row * LDT + kb);
      bLo[ni] = ld_frag_b(BsLo[buf] + row * LDT + kb);
    }

    // ---- 3-product split-precision WMMA (24 x v_wmma per iter per wave) ----
#pragma unroll
    for (int mi = 0; mi < 2; ++mi)
#pragma unroll
      for (int ni = 0; ni < 4; ++ni) {
        acc[mi][ni] = wmma_bf16(aHi[mi], bHi[ni], acc[mi][ni]);
        acc[mi][ni] = wmma_bf16(aHi[mi], bLo[ni], acc[mi][ni]);
        acc[mi][ni] = wmma_bf16(aLo[mi], bHi[ni], acc[mi][ni]);
      }

    // Convert + store next tile into the other buffer, then one barrier.
    if (kt + 1 < NT) {
      stage(buf ^ 1);
      __syncthreads();
    }
  }

  // ---- epilogue: scale by 1/(||x|| ||w||) and store ----
  const int hh = lane >> 4;             // C/D layout: lanes 16-31 hold M+8
#pragma unroll
  for (int mi = 0; mi < 2; ++mi) {
    const int rbase = m0 + wm * 32 + mi * 16 + hh * 8;
    float xr[8];
#pragma unroll
    for (int j = 0; j < 8; ++j) xr[j] = xrn[rbase + j];
#pragma unroll
    for (int ni = 0; ni < 4; ++ni) {
      const int c = n0 + wn * 64 + ni * 16 + fr;
      if (c < NCLS) {
        const float wv = wrn[c];
#pragma unroll
        for (int j = 0; j < 8; ++j)
          out[(size_t)(rbase + j) * NCLS + c] = acc[mi][ni][j] * xr[j] * wv;
      }
    }
  }
}

// ---------------------------------------------------------------------------
// Launch
// ---------------------------------------------------------------------------
extern "C" void kernel_launch(void* const* d_in, const int* in_sizes, int n_in,
                              void* d_out, int out_size, void* d_ws, size_t ws_size,
                              hipStream_t stream) {
  const float* x = (const float*)d_in[0];   // [32768, 512]
  const float* w = (const float*)d_in[1];   // [1000, 512]
  float* out = (float*)d_out;               // [32768, 1000]

  float* xrn = (float*)d_ws;                // [32768] 1/||x_n||
  float* wrn = xrn + BATCH;                 // [1000]  1/||w_c||

  cosine_rnorm_kernel<<<BATCH / 8, 256, 0, stream>>>(x, xrn, BATCH);
  cosine_rnorm_kernel<<<(NCLS + 7) / 8, 256, 0, stream>>>(w, wrn, NCLS);

  dim3 grid(BATCH / BM, (NCLS + BN - 1) / BN);  // 256 x 8 blocks
  cosine_gemm_kernel<<<grid, 256, 0, stream>>>(x, w, xrn, wrn, out);
}
